// Fp8DeepGEMMLinear_28432683499582
// MI455X (gfx1250) — 8-bit matrix operand order repaired and checked by reading, compiled, not run on hardware
//
#include <hip/hip_runtime.h>
#include <hip/hip_bf16.h>

#define GROUP   128
#define FP8_MAX 448.0f
#define EPS_Q   1e-4f

#define KDIM 4096
#define NDIM 4096
#define KG   (KDIM / GROUP)

typedef int   v16i __attribute__((ext_vector_type(16)));
typedef float v8f  __attribute__((ext_vector_type(8)));

__device__ __forceinline__ unsigned char f32_to_e4m3_byte(float x) {
    unsigned int u  = __float_as_uint(x);
    unsigned char s = (unsigned char)((u >> 24) & 0x80u);
    float a = fabsf(x);
    if (a >= FP8_MAX) return (unsigned char)(s | 0x7E);
    if (a < 0.0009765625f) return s;
    int e; float fr = frexpf(a, &e);
    int E = e + 6;
    if (E >= 1) {
        unsigned int m = (unsigned int)rintf((fr * 2.0f - 1.0f) * 8.0f);
        if (m == 8u) { m = 0u; ++E; }
        if (E > 15 || (E == 15 && m > 6u)) return (unsigned char)(s | 0x7E);
        return (unsigned char)(s | (unsigned char)((E << 3) | m));
    } else {
        unsigned int m = (unsigned int)rintf(a * 512.0f);
        if (m >= 8u) return (unsigned char)(s | 0x08);
        return (unsigned char)(s | (unsigned char)m);
    }
}

__device__ __forceinline__ unsigned int pack4_e4m3(float x0, float x1, float x2, float x3) {
#if defined(__AMDGCN__) && __has_builtin(__builtin_amdgcn_cvt_pk_fp8_f32)
    int p = 0;
    p = __builtin_amdgcn_cvt_pk_fp8_f32(x0, x1, p, false);
    p = __builtin_amdgcn_cvt_pk_fp8_f32(x2, x3, p, true);
    return (unsigned int)p;
#else
    return (unsigned int)f32_to_e4m3_byte(x0)
         | ((unsigned int)f32_to_e4m3_byte(x1) << 8)
         | ((unsigned int)f32_to_e4m3_byte(x2) << 16)
         | ((unsigned int)f32_to_e4m3_byte(x3) << 24);
#endif
}

__global__ void quant_act_kernel(const __hip_bfloat16* __restrict__ A,
                                 unsigned char* __restrict__ Aq,
                                 float* __restrict__ As,
                                 int M) {
    const int gid  = blockIdx.x * 8 + (threadIdx.x >> 5);
    const int lane = threadIdx.x & 31;
    const int m    = gid >> 5;
    const int kg   = gid & (KG - 1);
    if (m >= M) return;

    const size_t base = (size_t)m * KDIM + (size_t)kg * GROUP + (size_t)lane * 4;
    uint2 d = *(const uint2*)((const unsigned short*)A + base);
    float x0 = __uint_as_float(d.x << 16);
    float x1 = __uint_as_float(d.x & 0xFFFF0000u);
    float x2 = __uint_as_float(d.y << 16);
    float x3 = __uint_as_float(d.y & 0xFFFF0000u);

    float amax = fmaxf(fmaxf(fabsf(x0), fabsf(x1)), fmaxf(fabsf(x2), fabsf(x3)));
    #pragma unroll
    for (int off = 16; off > 0; off >>= 1)
        amax = fmaxf(amax, __shfl_xor(amax, off, 32));

    const float scale = fmaxf(amax, EPS_Q) / FP8_MAX;
    const float inv   = 1.0f / scale;

    float q0 = fminf(fmaxf(x0 * inv, -FP8_MAX), FP8_MAX);
    float q1 = fminf(fmaxf(x1 * inv, -FP8_MAX), FP8_MAX);
    float q2 = fminf(fmaxf(x2 * inv, -FP8_MAX), FP8_MAX);
    float q3 = fminf(fmaxf(x3 * inv, -FP8_MAX), FP8_MAX);

    *(unsigned int*)(Aq + base) = pack4_e4m3(q0, q1, q2, q3);
    if (lane == 0) As[(size_t)m * KG + kg] = scale;
}

__global__ void quant_wt_kernel(const float* __restrict__ W,
                                unsigned char* __restrict__ Wq) {
    const long long i = ((long long)blockIdx.x * 256 + threadIdx.x) * 4;
    float4 w = *(const float4*)(W + i);
    *(unsigned int*)(Wq + i) = pack4_e4m3(w.x, w.y, w.z, w.w);
}

union FragU {
    v16i v;
    unsigned long long u64[8];
    uint4 u128[4];
};
union F4 { float4 v; float f[4]; };

__global__ __launch_bounds__(256)
void gemm_fp8_kernel(const unsigned char* __restrict__ Aq,
                     const float* __restrict__ As,
                     const unsigned char* __restrict__ Wq,
                     const float* __restrict__ Ws,
                     __hip_bfloat16* __restrict__ out,
                     int M) {
    __shared__ float sTab[KG * 64];

    const int tid  = threadIdx.x;
    const int lane = tid & 31;
    const int wave = tid >> 5;
    const int wm   = wave >> 2;
    const int wn   = wave & 3;
    const int half = lane >> 4;
    const int l16  = lane & 15;

    const int mBlk = blockIdx.y * 64;
    const int m0 = mBlk + wm * 32;
    const int n0 = blockIdx.x * 128 + wn * 32;

    {
        const float* wsCol = Ws + (size_t)blockIdx.x * KG;
        #pragma unroll
        for (int i = 0; i < 8; ++i) {
            const int idx = tid + i * 256;
            const int kg  = idx >> 6;
            const int row = idx & 63;
            sTab[idx] = As[(size_t)(mBlk + row) * KG + kg] * wsCol[kg];
        }
    }
    __syncthreads();

    const unsigned char* aRow0 = Aq + (size_t)(m0 + 0  + l16) * KDIM + half * 16;
    const unsigned char* aRow1 = Aq + (size_t)(m0 + 16 + l16) * KDIM + half * 16;
    const unsigned char* bRow0 = Wq + (size_t)(n0 + 0  + l16) * KDIM + half * 16;
    const unsigned char* bRow1 = Wq + (size_t)(n0 + 16 + l16) * KDIM + half * 16;
    const float* sRow = sTab + wm * 32 + half * 8;

    FragU fa00, fa01, fb00, fb01;
    FragU fa10, fa11, fb10, fb11;

    v8f acc00 = {}, acc01 = {}, acc10 = {}, acc11 = {};

    auto loadA = [&](FragU& f, const unsigned char* row, int kb) {
        #pragma unroll
        for (int j = 0; j < 4; ++j) {
            f.u128[j] = *(const uint4*)(row + kb + 32 * j);
        }
    };
    auto loadB = [&](FragU& f, const unsigned char* row, int kb) {
        #pragma unroll
        for (int g = 0; g < 4; ++g)
            f.u128[g] = *(const uint4*)(row + kb + 32 * g);
    };

    auto step = [&](int kg, FragU& fa0, FragU& fa1, FragU& fb0, FragU& fb1) {
        const float* sp = sRow + kg * 64;
        F4 s00, s01, s10, s11;
        s00.v = *(const float4*)(sp + 0);
        s01.v = *(const float4*)(sp + 4);
        s10.v = *(const float4*)(sp + 16);
        s11.v = *(const float4*)(sp + 20);

        const v8f cz = {0.f, 0.f, 0.f, 0.f, 0.f, 0.f, 0.f, 0.f};
        v8f c0 = __builtin_amdgcn_wmma_f32_16x16x128_fp8_fp8(fa0.v, fb0.v, (short)0, cz, false, false);
        v8f c1 = __builtin_amdgcn_wmma_f32_16x16x128_fp8_fp8(fa0.v, fb1.v, (short)0, cz, false, false);
        v8f c2 = __builtin_amdgcn_wmma_f32_16x16x128_fp8_fp8(fa1.v, fb0.v, (short)0, cz, false, false);
        v8f c3 = __builtin_amdgcn_wmma_f32_16x16x128_fp8_fp8(fa1.v, fb1.v, (short)0, cz, false, false);
        #pragma unroll
        for (int r = 0; r < 8; ++r) {
            const float sm0 = (r < 4) ? s00.f[r] : s01.f[r - 4];
            const float sm1 = (r < 4) ? s10.f[r] : s11.f[r - 4];
            acc00[r] += c0[r] * sm0;
            acc01[r] += c1[r] * sm0;
            acc10[r] += c2[r] * sm1;
            acc11[r] += c3[r] * sm1;
        }
    };

    loadA(fa00, aRow0, 0); loadA(fa01, aRow1, 0);
    loadB(fb00, bRow0, 0); loadB(fb01, bRow1, 0);

    for (int kg = 0; kg < KG; kg += 2) {
        const int kb1 = (kg + 1) << 7;
        loadA(fa10, aRow0, kb1); loadA(fa11, aRow1, kb1);
        loadB(fb10, bRow0, kb1); loadB(fb11, bRow1, kb1);

        step(kg, fa00, fa01, fb00, fb01);

        if (kg + 2 < KG) {
            const int kb2 = (kg + 2) << 7;
            loadA(fa00, aRow0, kb2); loadA(fa01, aRow1, kb2);
            loadB(fb00, bRow0, kb2); loadB(fb01, bRow1, kb2);
        }

        step(kg + 1, fa10, fa11, fb10, fb11);
    }

    auto storeTile = [&](const v8f& accv, int mt, int nt) {
        #pragma unroll
        for (int r = 0; r < 8; ++r) {
            const int row = m0 + mt * 16 + r + half * 8;
            const int col = n0 + nt * 16 + l16;
            if (row < M)
                out[(size_t)row * NDIM + col] = __float2bfloat16(accv[r]);
        }
    };
    storeTile(acc00, 0, 0);
    storeTile(acc01, 0, 1);
    storeTile(acc10, 1, 0);
    storeTile(acc11, 1, 1);
}

extern "C" void kernel_launch(void* const* d_in, const int* in_sizes, int n_in,
                              void* d_out, int out_size, void* d_ws, size_t ws_size,
                              hipStream_t stream) {
    const int M = in_sizes[0] / KDIM;

    const __hip_bfloat16* A  = (const __hip_bfloat16*)d_in[0];
    const float*          W  = (const float*)d_in[1];
    const float*          Ws = (const float*)d_in[2];
    __hip_bfloat16*       O  = (__hip_bfloat16*)d_out;

    unsigned char* base = (unsigned char*)d_ws;
    size_t aq_bytes = ((size_t)M * KDIM + 255) & ~(size_t)255;
    size_t wq_bytes = ((size_t)NDIM * KDIM + 255) & ~(size_t)255;
    unsigned char* Aq = base;
    unsigned char* Wq = base + aq_bytes;
    float*         As = (float*)(base + aq_bytes + wq_bytes);
    (void)ws_size; (void)n_in; (void)out_size;

    const int groups = M * KG;
    quant_act_kernel<<<(groups + 7) / 8, 256, 0, stream>>>(A, Aq, As, M);

    const long long wElems = (long long)NDIM * KDIM;
    quant_wt_kernel<<<(unsigned int)(wElems / 4 / 256), 256, 0, stream>>>(W, Wq);

    dim3 grid(NDIM / 128, (M + 63) / 64);
    gemm_fp8_kernel<<<grid, 256, 0, stream>>>(Aq, As, Wq, Ws, O, M);
}
